// CI4GI_2783138808496
// MI455X (gfx1250) — matrix operand order repaired and checked by reading, not compiled, not run on hardware
//
#include <hip/hip_runtime.h>
#include <hip/hip_bf16.h>

typedef __attribute__((ext_vector_type(16))) _Float16 v16h;
typedef __attribute__((ext_vector_type(8)))  float    v8f;
typedef __attribute__((ext_vector_type(4)))  float    v4f;

#define N_U 4096
#define N_I 4096
#define N_G 1024
#define N_D 64

// ---------------------------------------------------------------------------
// Pack B (K x 64, fp32, optionally a row-concat of two sources) into the
// contiguous-k B-fragment order this file uses for v_wmma_f32_16x16x32_f16:
//   per 16x32 B tile: lanes 0-15 hold K=0..15 (col = lane), lanes 16-31 hold
//   K=16..31.  Pack order: [(kb32*4 + ntile)*32 + lane] * 16 halves.
// One thread produces one lane's 16 contiguous halves (32B store).
// ---------------------------------------------------------------------------
__global__ void pack_b_kernel(const float* __restrict__ src1, int n1,
                              const float* __restrict__ src2,
                              _Float16* __restrict__ bpack, int K) {
  int t = blockIdx.x * blockDim.x + threadIdx.x;
  int total = (K >> 5) * 4 * 32;
  if (t >= total) return;
  int lane = t & 31;
  int nt   = (t >> 5) & 3;
  int kb32 = t >> 7;
  int grp  = lane >> 4;
  int col  = nt * 16 + (lane & 15);
  int k0   = kb32 * 32 + grp * 16;
  v16h out;
#pragma unroll
  for (int j = 0; j < 16; ++j) {
    int k = k0 + j;
    float f = (k < n1) ? src1[(size_t)k * N_D + col]
                       : src2[(size_t)(k - n1) * N_D + col];
    out[j] = (_Float16)f;
  }
  ((v16h*)bpack)[t] = out;
}

// ---------------------------------------------------------------------------
// C[M,64] += A[M,K] (fp32, converted to f16 on the fly) * Bpack (f16 frags)
// block = 256 threads = 8 waves; wave w owns rows [blk*128 + w*16, +16).
// grid.y = K-split; partial sums combined with global_atomic_add_f32
// (C must be zeroed beforehand).
// A-fragment layout (16-bit A 16x32): lane l holds row l&15;
//   lanes 0-15:  K = kb+{0..15}  (halves 0-15)
//   lanes 16-31: K = kb+{16..31} (halves 0-15)
// ---------------------------------------------------------------------------
__global__ void wmma_gemm_n64_kernel(const float* __restrict__ A,
                                     const _Float16* __restrict__ bpack,
                                     float* __restrict__ C, int K) {
  const int lane = threadIdx.x & 31;
  const int wave = threadIdx.x >> 5;
  const int grp  = lane >> 4;
  const int n    = lane & 15;
  const int m0   = blockIdx.x * 128 + wave * 16;
  const int kPer = K / gridDim.y;              // multiple of 32 by launch
  const int kBeg = blockIdx.y * kPer;
  const int kEnd = kBeg + kPer;

  v8f acc0 = {}, acc1 = {}, acc2 = {}, acc3 = {};
  const float* arow  = A + (size_t)(m0 + n) * K;
  const v16h*  bbase = (const v16h*)bpack + lane;

  for (int kb = kBeg; kb < kEnd; kb += 32) {
    __builtin_prefetch((const void*)(arow + kb + 256), 0, 1);
    const float* ap = arow + kb + grp * 16;
    v4f a0 = *(const v4f*)(ap);
    v4f a1 = *(const v4f*)(ap + 4);
    v4f a2 = *(const v4f*)(ap + 8);
    v4f a3 = *(const v4f*)(ap + 12);
    v16h a;
#pragma unroll
    for (int j = 0; j < 4; ++j) {
      a[j]      = (_Float16)a0[j];
      a[4 + j]  = (_Float16)a1[j];
      a[8 + j]  = (_Float16)a2[j];
      a[12 + j] = (_Float16)a3[j];
    }
    const v16h* bp = bbase + (size_t)(kb >> 5) * 128;
    v16h b0 = bp[0];
    v16h b1 = bp[32];
    v16h b2 = bp[64];
    v16h b3 = bp[96];
    acc0 = __builtin_amdgcn_wmma_f32_16x16x32_f16(false, a, false, b0,
                                                  (short)0, acc0, false, false);
    acc1 = __builtin_amdgcn_wmma_f32_16x16x32_f16(false, a, false, b1,
                                                  (short)0, acc1, false, false);
    acc2 = __builtin_amdgcn_wmma_f32_16x16x32_f16(false, a, false, b2,
                                                  (short)0, acc2, false, false);
    acc3 = __builtin_amdgcn_wmma_f32_16x16x32_f16(false, a, false, b3,
                                                  (short)0, acc3, false, false);
  }

  // C/D layout: VGPR v -> row m0 + grp*8 + v, col n (+ ntile*16)
#pragma unroll
  for (int v = 0; v < 8; ++v) {
    float* cr = C + (size_t)(m0 + grp * 8 + v) * N_D + n;
    atomicAdd(cr + 0,  acc0[v]);
    atomicAdd(cr + 16, acc1[v]);
    atomicAdd(cr + 32, acc2[v]);
    atomicAdd(cr + 48, acc3[v]);
  }
}

// ---------------------------------------------------------------------------
// One wave per row: out[r] = sum_c A[r,c]          (rowsum)
//                   out[r] = sum_c A[r,c] * x[c]   (matvec)
// ---------------------------------------------------------------------------
__global__ void rowsum_kernel(const float* __restrict__ A,
                              float* __restrict__ out, int rows, int cols) {
  int w = (blockIdx.x * blockDim.x + threadIdx.x) >> 5;
  int lane = threadIdx.x & 31;
  if (w >= rows) return;
  const float* r = A + (size_t)w * cols;
  float s = 0.f;
  for (int c = lane; c < cols; c += 32) s += r[c];
#pragma unroll
  for (int off = 16; off; off >>= 1) s += __shfl_xor(s, off, 32);
  if (lane == 0) out[w] = s;
}

__global__ void matvec_kernel(const float* __restrict__ A,
                              const float* __restrict__ x,
                              float* __restrict__ out, int rows, int cols) {
  int w = (blockIdx.x * blockDim.x + threadIdx.x) >> 5;
  int lane = threadIdx.x & 31;
  if (w >= rows) return;
  const float* r = A + (size_t)w * cols;
  float s = 0.f;
  for (int c = lane; c < cols; c += 32) s += r[c] * x[c];
#pragma unroll
  for (int off = 16; off; off >>= 1) s += __shfl_xor(s, off, 32);
  if (lane == 0) out[w] = s;
}

// ---------------------------------------------------------------------------
// Assemble d_out = [final_u_emb (4096x128) ; final_g_emb (1024x128)]
// ---------------------------------------------------------------------------
__global__ void finalize_kernel(const float* __restrict__ ug2,
                                const float* __restrict__ ig2,
                                const float* __restrict__ uei,
                                const float* __restrict__ t2,
                                const float* __restrict__ counts,
                                float* __restrict__ out) {
  int idx = blockIdx.x * blockDim.x + threadIdx.x;
  if (idx >= (N_U + N_G) * 128) return;
  int row = idx >> 7;
  int c   = idx & 127;
  float v;
  if (row < N_U) {
    if (c < N_D) {
      v = ug2[(size_t)row * N_D + c];                       // user_emb
    } else {
      int cc = c - N_D;
      float cnt = counts[row];
      if (cnt == 0.f) cnt = 1.f;
      v = uei[(size_t)row * N_D + cc] +
          0.1f * (t2[(size_t)row * N_D + cc] / cnt);        // user_emb_i_side
    }
  } else {
    int g = row - N_U;
    v = (c < N_D) ? ug2[(size_t)(N_U + g) * N_D + c]        // group_u_side
                  : ig2[(size_t)(N_U + g) * N_D + (c - N_D)]; // group_i_side
  }
  out[idx] = v;
}

// ---------------------------------------------------------------------------
extern "C" void kernel_launch(void* const* d_in, const int* in_sizes, int n_in,
                              void* d_out, int out_size, void* d_ws,
                              size_t ws_size, hipStream_t stream) {
  (void)in_sizes; (void)n_in; (void)out_size; (void)ws_size;
  const float* uei     = (const float*)d_in[0];  // user_emb_interest
  const float* ued     = (const float*)d_in[1];  // user_emb_distinct
  const float* item    = (const float*)d_in[2];  // item_emb
  const float* gei     = (const float*)d_in[3];  // group_emb_interest
  const float* ged     = (const float*)d_in[4];  // group_emb_distinct
  const float* item_hg = (const float*)d_in[5];  // [5120,5120]
  const float* user_hg = (const float*)d_in[6];  // [5120,5120]
  const float* u_ssl   = (const float*)d_in[8];  // [4096,1024]
  const float* gi_ssl  = (const float*)d_in[9];  // [1024,4096]
  // d_in[7] adj, d_in[10..12] gat_a/fc1_* are dead code in the reference.

  const int KIG = N_I + N_G;  // 5120
  char* ws = (char*)d_ws;
  size_t off = 0;
  auto alloc = [&](size_t bytes) -> void* {
    void* p = ws + off;
    off = (off + bytes + 255) & ~(size_t)255;
    return p;
  };
  float*     T     = (float*)alloc((size_t)KIG * N_D * 4);   // layer-1 temp
  float*     IG2   = (float*)alloc((size_t)KIG * N_D * 4);
  float*     UG2   = (float*)alloc((size_t)KIG * N_D * 4);
  float*     T1    = (float*)alloc((size_t)N_G * N_D * 4);   // gi_ssl @ tmp_item
  float*     T2    = (float*)alloc((size_t)N_U * N_D * 4);   // u_ssl @ T1
  float*     GROWS = (float*)alloc((size_t)N_G * 4);
  float*     CNTS  = (float*)alloc((size_t)N_U * 4);
  _Float16*  BP    = (_Float16*)alloc((size_t)KIG * N_D * 2);

  auto packGrid = [](int K) { return dim3((K * 4 + 255) / 256); };

  // ---- chain 1: ig2 = item_hg @ (item_hg @ [item_emb; group_emb_interest])
  pack_b_kernel<<<packGrid(KIG), 256, 0, stream>>>(item, N_I, gei, BP, KIG);
  hipMemsetAsync(T, 0, (size_t)KIG * N_D * 4, stream);
  wmma_gemm_n64_kernel<<<dim3(KIG / 128, 4), 256, 0, stream>>>(item_hg, BP, T, KIG);
  pack_b_kernel<<<packGrid(KIG), 256, 0, stream>>>(T, KIG, nullptr, BP, KIG);
  hipMemsetAsync(IG2, 0, (size_t)KIG * N_D * 4, stream);
  wmma_gemm_n64_kernel<<<dim3(KIG / 128, 4), 256, 0, stream>>>(item_hg, BP, IG2, KIG);

  // ---- chain 2: ug2 = user_hg @ (user_hg @ [user_emb_distinct; group_emb_distinct])
  pack_b_kernel<<<packGrid(KIG), 256, 0, stream>>>(ued, N_U, ged, BP, KIG);
  hipMemsetAsync(T, 0, (size_t)KIG * N_D * 4, stream);
  wmma_gemm_n64_kernel<<<dim3(KIG / 128, 4), 256, 0, stream>>>(user_hg, BP, T, KIG);
  pack_b_kernel<<<packGrid(KIG), 256, 0, stream>>>(T, KIG, nullptr, BP, KIG);
  hipMemsetAsync(UG2, 0, (size_t)KIG * N_D * 4, stream);
  wmma_gemm_n64_kernel<<<dim3(KIG / 128, 4), 256, 0, stream>>>(user_hg, BP, UG2, KIG);

  // ---- ssl chain, re-associated:
  // T1 = gi_ssl @ tmp_item   (tmp_item = IG2 rows [0, 4096))
  pack_b_kernel<<<packGrid(N_I), 256, 0, stream>>>(IG2, N_I, nullptr, BP, N_I);
  hipMemsetAsync(T1, 0, (size_t)N_G * N_D * 4, stream);
  wmma_gemm_n64_kernel<<<dim3(N_G / 128, 8), 256, 0, stream>>>(gi_ssl, BP, T1, N_I);
  // T2 = u_ssl @ T1
  pack_b_kernel<<<packGrid(N_G), 256, 0, stream>>>(T1, N_G, nullptr, BP, N_G);
  hipMemsetAsync(T2, 0, (size_t)N_U * N_D * 4, stream);
  wmma_gemm_n64_kernel<<<dim3(N_U / 128, 4), 256, 0, stream>>>(u_ssl, BP, T2, N_G);
  // counts = u_ssl @ rowsum(gi_ssl)   (exact: non-negative integer sums)
  rowsum_kernel<<<(N_G * 32 + 255) / 256, 256, 0, stream>>>(gi_ssl, GROWS, N_G, N_I);
  matvec_kernel<<<(N_U * 32 + 255) / 256, 256, 0, stream>>>(u_ssl, GROWS, CNTS, N_U, N_G);

  // ---- assemble outputs
  finalize_kernel<<<((N_U + N_G) * 128 + 255) / 256, 256, 0, stream>>>(
      UG2, IG2, uei, T2, CNTS, (float*)d_out);
}
